// _multiCodebookQuantization_79413945303132
// MI455X (gfx1250) — compile-verified, matrix operand order checked by reading, not run on hardware
//
#include <hip/hip_runtime.h>
#include <hip/hip_bf16.h>

typedef __attribute__((ext_vector_type(16))) _Float16 v16h;
typedef __attribute__((ext_vector_type(8)))  float    v8f;

#define KCODES 4096
#define DDIM   32

// One block: 16 rows (n,hw positions) of one codebook m, all K=4096 codes.
// 8 waves * (16 codes/tile) * 32 iters = 4096 codes.
__global__ __launch_bounds__(256)
void _mcq_fused_kernel(const float* __restrict__ x,
                       const float* __restrict__ codebook,
                       const float* __restrict__ freqEMA,
                       const float* __restrict__ temperature,
                       const float* __restrict__ dropn,
                       const float* __restrict__ gumb,
                       float* __restrict__ samp,
                       float* __restrict__ codep,
                       float* __restrict__ oneh,
                       float* __restrict__ logp)
{
    __shared__ float s_x[16][DDIM];
    __shared__ float s_x2[16];
    __shared__ int   s_cnt;
    __shared__ float s_lmax[8][16];
    __shared__ int   s_lidx[8][16];
    __shared__ float s_gmax[8][16];
    __shared__ float s_gsum[8][16];
    __shared__ int   s_gidx[8][16];

    const int tid     = threadIdx.x;
    const int lane    = tid & 31;
    const int wave    = tid >> 5;
    const int m       = blockIdx.x & 3;
    const int rowTile = blockIdx.x >> 2;

    // ---- expo scalar: mean(freqEMA > EPS) over all M*K (L2-resident, redundant per block)
    if (tid == 0) s_cnt = 0;
    __syncthreads();
    {
        int c = 0;
        for (int j = tid; j < 4 * KCODES; j += 256)
            c += (freqEMA[j] > 1e-7f) ? 1 : 0;
        atomicAdd(&s_cnt, c);
    }

    // ---- stage x tile (16 rows x 32 dims) into LDS
    for (int t = tid; t < 16 * DDIM; t += 256) {
        int i = t >> 5, d = t & 31;
        int r = rowTile * 16 + i;
        int n = r >> 10, hw = r & 1023;
        s_x[i][d] = x[(size_t)(n * 128 + m * 32 + d) * 1024 + hw];
    }
    __syncthreads();
    if (tid < 16) {
        float s2 = 0.f;
        for (int d = 0; d < DDIM; ++d) { float v = s_x[tid][d]; s2 += v * v; }
        s_x2[tid] = s2;
    }
    __syncthreads();

    float cu = (float)s_cnt * (1.0f / 16384.0f);
    cu = fminf(fmaxf(cu, 0.f), 1.f);
    const float expo     = -11.0f * cu * cu + 12.0f;           // -(BITS-1)*cu^2 + BITS
    const float rcp_expo = 1.0f / expo;                        // expo in [1,12] > 0

    const float t_     = fmaxf(temperature[m], 1e-7f);         // lower_bound
    const float negT64 = -t_ * (1.0f / 64.0f);                 // logit = dist * (-t/SCALE)

    const int hi    = lane >> 4;       // lane half (selects K sub-ranges)
    const int col   = lane & 15;       // N (code) within tile
    const int rbase = hi * 8;          // C/D rows held by this lane: rbase..rbase+7

    // ---- A fragment (16x32 f16), loop-invariant. Doc layout:
    // lane<16: M=lane, K = j<8 ? j : j+8 ; lane>=16: same +8
    v16h a;
    {
        int mrow = lane & 15;
#pragma unroll
        for (int j = 0; j < 16; ++j) {
            int d = (j < 8 ? j : j + 8) + hi * 8;
            a[j] = (_Float16)s_x[mrow][d];
        }
    }

    float    x2r[8];
    unsigned base[8];
#pragma unroll
    for (int i = 0; i < 8; ++i) {
        x2r[i] = s_x2[rbase + i];
        int r = rowTile * 16 + rbase + i;
        int n = r >> 10, hw = r & 1023;
        base[i] = (unsigned)((n * 4 + m) * 1024 + hw) * (unsigned)KCODES;
    }

    float lmax[8], gmax[8], gsum[8];
    int   lidx[8], gidx[8];
#pragma unroll
    for (int i = 0; i < 8; ++i) {
        lmax[i] = -3.0e38f; lidx[i] = 0;
        gmax[i] = -3.0e38f; gsum[i] = 0.f; gidx[i] = 0;
    }

    const float* cbm = codebook + (size_t)m * KCODES * DDIM;
    const float* fem = freqEMA + m * KCODES;

    for (int kt = 0; kt < 32; ++kt) {
        const int k0 = kt * 128 + wave * 16;
        const int k  = k0 + col;

        // ---- B fragment (32x16 f16) + per-column |c|^2 partial (f32)
        float c2p = 0.f;
        v16h  b;
        const float* cb = cbm + (size_t)k * DDIM + hi * 8;
#pragma unroll
        for (int j = 0; j < 8; ++j)  { float v = cb[j];     c2p += v * v; b[j] = (_Float16)v; }
#pragma unroll
        for (int j = 8; j < 16; ++j) { float v = cb[j + 8]; c2p += v * v; b[j] = (_Float16)v; }
        float c2 = c2p + __shfl_xor(c2p, 16, 32);

        v8f acc = {};
        acc = __builtin_amdgcn_wmma_f32_16x16x32_f16(
            false, a, false, b, (short)0, acc, false, false);

        // drop test in log2 domain: dn^expo < freq  <=>  log2(dn) < log2(freq)/expo
        const float thr = __log2f(fem[k]) * rcp_expo;          // per-k, shared by 8 rows

#pragma unroll
        for (int i = 0; i < 8; ++i) {
            size_t idx = (size_t)base[i] + (unsigned)k;
            float  lg  = (x2r[i] + c2 - 2.0f * acc[i]) * negT64;
            float  dn  = __builtin_nontemporal_load(&dropn[idx]);
            float  gn  = __builtin_nontemporal_load(&gumb[idx]);
            lg += (__log2f(dn) < thr) ? -1e9f : 0.0f;          // random drop (branchless)
            __builtin_nontemporal_store(lg,   &logp[idx]);
            __builtin_nontemporal_store(0.0f, &samp[idx]);     // sample == 0 off-argmax (exact)
            __builtin_nontemporal_store(0.0f, &oneh[idx]);

            // argmax(logit) — first-occurrence (k strictly increasing per lane)
            bool bl = lg > lmax[i];
            lidx[i] = bl ? k  : lidx[i];
            lmax[i] = bl ? lg : lmax[i];

            // online softmax for g = logit + gumbel, branchless single-exp update:
            // exp(min(g,gmax) - max(g,gmax)) = exp(-|g-gmax|); other side contributes exp(0)=1
            float g  = lg + gn;
            bool  up = g > gmax[i];
            float e  = __expf(-fabsf(g - gmax[i]));
            gsum[i]  = up ? __builtin_fmaf(gsum[i], e, 1.0f) : (gsum[i] + e);
            gidx[i]  = up ? k : gidx[i];
            gmax[i]  = up ? g : gmax[i];
        }
    }

    // ---- butterfly reduce across the 16 lanes holding each row (bit4 preserved)
#pragma unroll
    for (int off = 8; off >= 1; off >>= 1) {
#pragma unroll
        for (int i = 0; i < 8; ++i) {
            float om = __shfl_xor(lmax[i], off, 32);
            int   oi = __shfl_xor(lidx[i], off, 32);
            if (om > lmax[i] || (om == lmax[i] && oi < lidx[i])) { lmax[i] = om; lidx[i] = oi; }
            float ogm = __shfl_xor(gmax[i], off, 32);
            float ogs = __shfl_xor(gsum[i], off, 32);
            int   ogi = __shfl_xor(gidx[i], off, 32);
            float nm  = fmaxf(gmax[i], ogm);
            gsum[i] = gsum[i] * __expf(gmax[i] - nm) + ogs * __expf(ogm - nm);
            if (ogm > gmax[i] || (ogm == gmax[i] && ogi < gidx[i])) gidx[i] = ogi;
            gmax[i] = nm;
        }
    }
    if (col == 0) {
#pragma unroll
        for (int i = 0; i < 8; ++i) {
            s_lmax[wave][rbase + i] = lmax[i]; s_lidx[wave][rbase + i] = lidx[i];
            s_gmax[wave][rbase + i] = gmax[i]; s_gsum[wave][rbase + i] = gsum[i];
            s_gidx[wave][rbase + i] = gidx[i];
        }
    }

    // Ensure all zero-stores have landed in L2 before the per-row fixups overwrite them.
    asm volatile("s_wait_storecnt 0" ::: "memory");
    __syncthreads();

    if (tid < 16) {
        int   row = tid;
        float LM = -3.0e38f; int LI = 0;
        float GM = -3.0e38f, GS = 0.f; int GI = 0;
        for (int w = 0; w < 8; ++w) {
            float om = s_lmax[w][row]; int oi = s_lidx[w][row];
            if (om > LM || (om == LM && oi < LI)) { LM = om; LI = oi; }
            float ogm = s_gmax[w][row], ogs = s_gsum[w][row]; int ogi = s_gidx[w][row];
            float nm  = fmaxf(GM, ogm);
            GS = GS * __expf(GM - nm) + ogs * __expf(ogm - nm);
            if (ogm > GM || (ogm == GM && ogi < GI)) GI = ogi;
            GM = nm;
        }
        int r = rowTile * 16 + row;
        int n = r >> 10, hw = r & 1023;
        unsigned rowg = (unsigned)((n * 4 + m) * 1024 + hw);
        codep[rowg] = (float)LI;                                 // argmax(logit)
        oneh[(size_t)rowg * KCODES + LI] = 1.0f;                 // one_hot
        float s = 1.0f / GS;                                     // softmax @ gumbel argmax
        samp[(size_t)rowg * KCODES + GI] = (1.0f + s) - s;       // y_hard + y_soft - y_soft
    }
}

extern "C" void kernel_launch(void* const* d_in, const int* in_sizes, int n_in,
                              void* d_out, int out_size, void* d_ws, size_t ws_size,
                              hipStream_t stream) {
    (void)in_sizes; (void)n_in; (void)out_size; (void)d_ws; (void)ws_size;
    const float* x  = (const float*)d_in[0];   // (4,128,32,32)
    const float* cb = (const float*)d_in[1];   // (4,4096,32)
    const float* fe = (const float*)d_in[2];   // (4,4096)
    const float* tp = (const float*)d_in[3];   // (4,1,1,1)
    const float* dn = (const float*)d_in[4];   // (4,4,32,32,4096)
    const float* gn = (const float*)d_in[5];   // (4,4,32,32,4096)

    float* out = (float*)d_out;
    const size_t BIG = 67108864ull;            // 4*4*32*32*4096
    float* samp  = out;                        // sample
    float* codep = samp + BIG;                 // code (16384)
    float* oneh  = codep + 16384;              // one_hot
    float* logp  = oneh + BIG;                 // logit

    _mcq_fused_kernel<<<dim3(1024), dim3(256), 0, stream>>>(
        x, cb, fe, tp, dn, gn, samp, codep, oneh, logp);
}
